// peepLSTM_76381698392870
// MI455X (gfx1250) — matrix operand order repaired and checked by reading, not compiled, not run on hardware
//
#include <hip/hip_runtime.h>
#include <hip/hip_bf16.h>

#define HID   1024
#define BATCH 256
#define SEQT  512
#define G4H   4096

typedef __bf16 v8bf  __attribute__((ext_vector_type(8)));
typedef __bf16 v16bf __attribute__((ext_vector_type(16)));
typedef float  v8f   __attribute__((ext_vector_type(8)));

static __device__ __forceinline__ unsigned short f2bf(float f) {
    unsigned int u = __float_as_uint(f);
    u += 0x7fffu + ((u >> 16) & 1u);          // round-to-nearest-even
    return (unsigned short)(u >> 16);
}
static __device__ __forceinline__ float sigf(float x) {
    return 1.0f / (1.0f + __expf(-x));
}

// ---------------------------------------------------------------------------
// Prep: G[v][g] = emb[v] . Wx[:,g] + bias[g], order [i | f | o | cand]
// ---------------------------------------------------------------------------
__global__ void prep_g(const float* __restrict__ emb,
                       const float* __restrict__ wix, const float* __restrict__ wfx,
                       const float* __restrict__ wox, const float* __restrict__ wcx,
                       const float* __restrict__ bi,  const float* __restrict__ bf,
                       const float* __restrict__ bo,  const float* __restrict__ bc,
                       float* __restrict__ G)
{
    int idx = blockIdx.x * blockDim.x + threadIdx.x;
    if (idx >= 3 * G4H) return;
    int v = idx >> 12;
    int g = idx & (G4H - 1);
    const float* e = emb + v * 512;
    const float* W; int col; float bias;
    if      (g < 1024) { W = wix; col = g;        bias = bi[col]; }
    else if (g < 2048) { W = wfx; col = g - 1024; bias = bf[col]; }
    else if (g < 3072) { W = wox; col = g - 2048; bias = bo[col]; }
    else               { W = wcx; col = g - 3072; bias = bc[col]; }
    float acc = bias;
    for (int k = 0; k < 512; ++k) acc += e[k] * W[k * HID + col];
    G[idx] = acc;
}

// ---------------------------------------------------------------------------
// Prep: bf16 copies of Wh.  Whif = [K=1024][N=2048] (i | f), Who = [1024][1024]
// ---------------------------------------------------------------------------
__global__ void prep_wh(const float* __restrict__ wih, const float* __restrict__ wfh,
                        const float* __restrict__ woh,
                        unsigned short* __restrict__ Whif, unsigned short* __restrict__ Who)
{
    int idx = blockIdx.x * blockDim.x + threadIdx.x;
    if (idx >= HID * 3072) return;
    int k = idx / 3072, n = idx % 3072;
    if      (n < 1024) Whif[k * 2048 + n] = f2bf(wih[k * HID + n]);
    else if (n < 2048) Whif[k * 2048 + n] = f2bf(wfh[k * HID + (n - 1024)]);
    else               Who [k * HID + (n - 2048)] = f2bf(woh[k * HID + (n - 2048)]);
}

__global__ void zero_c(float* __restrict__ cF, unsigned short* __restrict__ cB)
{
    int i = blockIdx.x * blockDim.x + threadIdx.x;
    if (i < BATCH * HID) { cF[i] = 0.0f; cB[i] = 0; }
}

// ---------------------------------------------------------------------------
// One recurrence step.  Per wave: one 16x16 output tile of gh for gates i and f
// (plus o at the last step), K=1024 via v_wmma_f32_16x16x32_bf16, then fused
// gate nonlinearities + c update in the D-fragment layout.
//   D layout: lane L, vgpr r -> M = m0 + r + (L>=16 ? 8:0), N = n0 + (L & 15)
// ---------------------------------------------------------------------------
template<bool LAST>
__global__ __launch_bounds__(256) void step_kernel(
    const unsigned short* __restrict__ cB,   // bf16 c (current)  [256][1024]
    const float*          __restrict__ cF,   // f32  c (current)
    unsigned short*       __restrict__ cBn,  // bf16 c (next)
    float*                __restrict__ cFn,  // f32  c (next)
    const unsigned short* __restrict__ Whif, // [1024][2048] bf16
    const unsigned short* __restrict__ Who,  // [1024][1024] bf16
    const float*          __restrict__ G,    // [3][4096]
    const int*            __restrict__ x,    // [256][512]
    float*                __restrict__ h,    // [256][1024] (written iff LAST)
    int t)
{
    const int lane   = threadIdx.x & 31;
    const int wave   = (blockIdx.x << 3) | (threadIdx.x >> 5);
    const int nwaves = gridDim.x << 3;

    for (int tile = wave; tile < 16 * 64; tile += nwaves) {
        const int mt = tile >> 6;          // 0..15  (batch tile)
        const int nt = tile & 63;          // 0..63  (hidden tile)
        const int m0 = mt << 4, n0 = nt << 4;

        v8f acc_i = {}; v8f acc_f = {}; v8f acc_o = {};

        // A fragment addressing (16-bit A 16x32 layout)
        const int am   = m0 + (lane & 15);
        const int asel = (lane >> 4) << 3;                 // lanes>=16 start at K+8
        const unsigned short* arow  = cB + (unsigned)am * HID;
        // B fragment addressing: lane&15 = N column, lanes>=16 start at K+8
        const unsigned short* browi = Whif + (unsigned)asel * 2048 + n0 + (lane & 15);
        const unsigned short* browf = browi + HID;
        const unsigned short* browo = Who  + (unsigned)asel * HID + n0 + (lane & 15);

        #pragma unroll 2
        for (int k0 = 0; k0 < HID; k0 += 32) {
            v8bf alo = *(const v8bf*)(arow + k0 + asel);
            v8bf ahi = *(const v8bf*)(arow + k0 + asel + 16);
            v16bf a = __builtin_shufflevector(alo, ahi,
                        0,1,2,3,4,5,6,7,8,9,10,11,12,13,14,15);

            const unsigned short* bi_ = browi + (unsigned)k0 * 2048;
            v16bf bI;
            for (int s = 0; s < 16; ++s) bI[s] = ((const __bf16*)bi_)[(s + (s & 8)) * 2048];
            acc_i = __builtin_amdgcn_wmma_f32_16x16x32_bf16(
                        false, a, false, bI, (short)0, acc_i, false, false);

            const unsigned short* bf_ = browf + (unsigned)k0 * 2048;
            v16bf bF;
            for (int s = 0; s < 16; ++s) bF[s] = ((const __bf16*)bf_)[(s + (s & 8)) * 2048];
            acc_f = __builtin_amdgcn_wmma_f32_16x16x32_bf16(
                        false, a, false, bF, (short)0, acc_f, false, false);

            if (LAST) {
                const unsigned short* bo_ = browo + (unsigned)k0 * HID;
                v16bf bO;
                for (int s = 0; s < 16; ++s) bO[s] = ((const __bf16*)bo_)[(s + (s & 8)) * HID];
                acc_o = __builtin_amdgcn_wmma_f32_16x16x32_bf16(
                            false, a, false, bO, (short)0, acc_o, false, false);
            }
        }

        // Fused epilogue: gates + c update (+ h at last step)
        const int nn    = n0 + (lane & 15);
        const int mbase = m0 + ((lane >> 4) << 3);
        #pragma unroll
        for (int r = 0; r < 8; ++r) {
            const int b = mbase + r;
            const int v = x[b * SEQT + t];
            const float* Gr = G + v * G4H;
            float gi   = Gr[nn]             + acc_i[r];
            float gf   = Gr[HID + nn]       + acc_f[r];
            float gc   = Gr[3 * HID + nn];
            float cold = cF[b * HID + nn];
            float cn   = sigf(gc) * sigf(gi) + cold * sigf(gf);
            cFn[b * HID + nn] = cn;
            cBn[b * HID + nn] = f2bf(cn);
            if (LAST) {
                float go  = Gr[2 * HID + nn] + acc_o[r];
                float pad = (float)((v + 1) >> 1);
                h[b * HID + nn] = tanhf(cn) * sigf(go) * pad;
            }
        }
    }
}

// ---------------------------------------------------------------------------
// Head: p[b][j] = h[b].w_ph[:,j] + b_p[j]; out = log_softmax over BATCH axis.
// One block per class j (10 blocks, 256 threads = one per batch row).
// ---------------------------------------------------------------------------
__global__ __launch_bounds__(256) void head_kernel(
    const float* __restrict__ h, const float* __restrict__ wph,
    const float* __restrict__ bp, float* __restrict__ out)
{
    __shared__ float red[256];
    const int j = blockIdx.x;       // class
    const int b = threadIdx.x;      // batch row
    const float* hr = h + b * HID;
    float acc = bp[j];
    for (int k = 0; k < HID; ++k) acc += hr[k] * wph[k * 10 + j];

    red[b] = acc; __syncthreads();
    for (int s = 128; s > 0; s >>= 1) {
        if (b < s) red[b] = fmaxf(red[b], red[b + s]);
        __syncthreads();
    }
    float mx = red[0]; __syncthreads();

    red[b] = __expf(acc - mx); __syncthreads();
    for (int s = 128; s > 0; s >>= 1) {
        if (b < s) red[b] += red[b + s];
        __syncthreads();
    }
    float lse = mx + __logf(red[0]);
    out[b * 10 + j] = acc - lse;
}

// ---------------------------------------------------------------------------
extern "C" void kernel_launch(void* const* d_in, const int* in_sizes, int n_in,
                              void* d_out, int out_size, void* d_ws, size_t ws_size,
                              hipStream_t stream)
{
    const int*   x   = (const int*)  d_in[0];
    const float* emb = (const float*)d_in[1];
    const float* wcx = (const float*)d_in[2];
    const float* bc  = (const float*)d_in[3];
    const float* wix = (const float*)d_in[4];
    const float* wih = (const float*)d_in[5];
    const float* bi  = (const float*)d_in[6];
    const float* wfx = (const float*)d_in[7];
    const float* wfh = (const float*)d_in[8];
    const float* bf  = (const float*)d_in[9];
    const float* wox = (const float*)d_in[10];
    const float* woh = (const float*)d_in[11];
    const float* bo  = (const float*)d_in[12];
    const float* wph = (const float*)d_in[13];
    const float* bp  = (const float*)d_in[14];
    (void)in_sizes; (void)n_in; (void)out_size; (void)ws_size;

    char* ws = (char*)d_ws;
    size_t off = 0;
    auto carve = [&](size_t bytes) -> char* {
        char* p = ws + off;
        off = (off + bytes + 255) & ~(size_t)255;
        return p;
    };
    float*          G    = (float*)         carve(3 * G4H * 4);
    unsigned short* Whif = (unsigned short*)carve((size_t)HID * 2048 * 2);
    unsigned short* Who  = (unsigned short*)carve((size_t)HID * HID * 2);
    float*          cF0  = (float*)         carve((size_t)BATCH * HID * 4);
    float*          cF1  = (float*)         carve((size_t)BATCH * HID * 4);
    unsigned short* cB0  = (unsigned short*)carve((size_t)BATCH * HID * 2);
    unsigned short* cB1  = (unsigned short*)carve((size_t)BATCH * HID * 2);
    float*          h    = (float*)         carve((size_t)BATCH * HID * 4);

    prep_g <<<48, 256, 0, stream>>>(emb, wix, wfx, wox, wcx, bi, bf, bo, bc, G);
    prep_wh<<<(HID * 3072 + 255) / 256, 256, 0, stream>>>(wih, wfh, woh, Whif, Who);
    zero_c <<<(BATCH * HID + 255) / 256, 256, 0, stream>>>(cF0, cB0);

    float*          cF[2] = {cF0, cF1};
    unsigned short* cB[2] = {cB0, cB1};

    for (int t = 0; t < 510; ++t) {
        const int cur = t & 1, nxt = cur ^ 1;
        step_kernel<false><<<128, 256, 0, stream>>>(
            cB[cur], cF[cur], cB[nxt], cF[nxt], Whif, Who, G, x, h, t);
    }
    {
        const int t = 510, cur = t & 1, nxt = cur ^ 1;
        step_kernel<true><<<128, 256, 0, stream>>>(
            cB[cur], cF[cur], cB[nxt], cF[nxt], Whif, Who, G, x, h, t);
    }
    head_kernel<<<10, 256, 0, stream>>>(h, wph, bp, (float*)d_out);
}
